// BehaviourGNNBlock_34067680592041
// MI455X (gfx1250) — matrix operand order repaired and checked by reading, not compiled, not run on hardware
//
#include <hip/hip_runtime.h>
#include <hip/hip_bf16.h>
#include <stdint.h>

// ---------------------------------------------------------------------------
// GraphSAGE(mean) x2 + BatchNorm + ReLU for MI455X (gfx1250, wave32).
//
// - All GEMM operands kept in f16 (activations pre-converted once per layer),
//   GEMMs on v_wmma_f32_16x16x32_f16 with f32 accumulation.
// - Each wave computes a 16x64 output strip: one A fragment feeds 4 WMMAs,
//   fragments are pure contiguous b128 loads (no VALU cvt in the hot loop).
// - Scatter-mean gathers f16 features (halves gather bytes) and accumulates
//   with f32 global atomics; whole working set (~116 MB) is L2-resident
//   (192 MB global L2), so atomics and tile re-reads resolve in L2.
// - Roofline: scatter traffic ~0.9 GB @ 23.3 TB/s dominates; 20 GFLOP of
//   GEMM is negligible on the WMMA path.
// ---------------------------------------------------------------------------

#define N_NODES 50000
#define N_EDGES 600000
#define IN_DIM  128
#define HID     256
#define BN_EPS  1e-5f

typedef _Float16 v16h __attribute__((ext_vector_type(16)));
typedef float    v8f  __attribute__((ext_vector_type(8)));

// ---------------- utility kernels ----------------

__global__ void zero_f32(float* __restrict__ p, long long n) {
  for (long long i = blockIdx.x * (long long)blockDim.x + threadIdx.x; i < n;
       i += (long long)gridDim.x * blockDim.x)
    p[i] = 0.0f;
}

__global__ void cvt_f32_to_f16(const float* __restrict__ s,
                               _Float16* __restrict__ d, int n) {
  for (int i = blockIdx.x * blockDim.x + threadIdx.x; i < n;
       i += gridDim.x * blockDim.x)
    d[i] = (_Float16)s[i];
}

// ---------------- graph aggregation ----------------

__global__ void count_deg(const int* __restrict__ dstv,
                          float* __restrict__ cnt, int E) {
  for (int e = blockIdx.x * blockDim.x + threadIdx.x; e < E;
       e += gridDim.x * blockDim.x)
    atomicAdd(&cnt[dstv[e]], 1.0f);
}

// sum[dst] += feat16[src]; 8 halves (one b128 load) per thread-item
__global__ void scatter_sum_f16(const _Float16* __restrict__ feat,
                                const int* __restrict__ srcv,
                                const int* __restrict__ dstv,
                                float* __restrict__ sum, int E, int d) {
  const int chunks = d >> 3;
  const long long total = (long long)E * chunks;
  for (long long idx = blockIdx.x * (long long)blockDim.x + threadIdx.x;
       idx < total; idx += (long long)gridDim.x * blockDim.x) {
    const int e = (int)(idx / chunks);
    const int c = (int)(idx - (long long)e * chunks);
    const int s = srcv[e], t = dstv[e];
    const _Float16* fp = feat + (size_t)s * d + c * 8;
    float* p = sum + (size_t)t * d + c * 8;
#pragma unroll
    for (int k = 0; k < 8; ++k) atomicAdd(p + k, (float)fp[k]);
  }
}

// mean16 = f16(sum / max(cnt, 1))  (fused divide + convert)
__global__ void mean_cvt(const float* __restrict__ sum,
                         const float* __restrict__ cnt,
                         _Float16* __restrict__ mean16, int n, int d) {
  const long long total = (long long)n * d;
  for (long long i = blockIdx.x * (long long)blockDim.x + threadIdx.x;
       i < total; i += (long long)gridDim.x * blockDim.x) {
    const int row = (int)(i / d);
    mean16[i] = (_Float16)(sum[i] / fmaxf(cnt[row], 1.0f));
  }
}

// ---------------- WMMA GEMM: out = Amean*Wl^T + Aroot*Wr^T + bias ----------
// Per-wave: 16 rows x 64 cols (4 accumulators share each A fragment).
// Fragment layouts (wave32):
//   A 16x32 f16: lane&15 = row M; halves {k0+half*16 .. +15}
//                -> one contiguous 32B chunk per lane (2x b128).
//   B 32x16 f16: lane&15 = col N; halves {k0+half*16 .. +15}
//                -> one contiguous 32B chunk per lane (2x b128).
//   C/D f32:     VGPR i -> row i + half*8; lane&15 -> col.
__global__ void sage_gemm_wmma(const _Float16* __restrict__ Amean,
                               const _Float16* __restrict__ Aroot,
                               const _Float16* __restrict__ Wl,
                               const _Float16* __restrict__ Wr,
                               const float* __restrict__ bias,
                               float* __restrict__ out, int n, int K) {
  const int lane = threadIdx.x & 31;
  const int wave = threadIdx.x >> 5;
  const int colGroups = HID / 64;                       // 4 strips of 64 cols
  const int totalWaves = (n / 16) * colGroups;          // n % 16 == 0
  const int w = blockIdx.x * (blockDim.x >> 5) + wave;
  if (w >= totalWaves) return;                          // wave-uniform branch
  const int cg = w & (colGroups - 1);
  const int tm = w / colGroups;
  const int m0 = tm * 16, n0 = cg * 64;
  const int half = lane >> 4;
  const int r = lane & 15;

  v8f acc0 = {}, acc1 = {}, acc2 = {}, acc3 = {};
  for (int pass = 0; pass < 2; ++pass) {
    const _Float16* __restrict__ arow =
        (pass ? Aroot : Amean) + (size_t)(m0 + r) * K;
    const _Float16* __restrict__ wbase = (pass ? Wr : Wl);
    for (int k0 = 0; k0 < K; k0 += 32) {
      v16h a;
      *(uint4*)&a       = *(const uint4*)(arow + k0 + half * 16);
      *((uint4*)&a + 1) = *(const uint4*)(arow + k0 + half * 16 + 8);
      v16h b0, b1, b2, b3;
      {
        const _Float16* w0 = wbase + (size_t)(n0 + 0 * 16 + r) * K + k0 + half * 16;
        const _Float16* w1 = wbase + (size_t)(n0 + 1 * 16 + r) * K + k0 + half * 16;
        const _Float16* w2 = wbase + (size_t)(n0 + 2 * 16 + r) * K + k0 + half * 16;
        const _Float16* w3 = wbase + (size_t)(n0 + 3 * 16 + r) * K + k0 + half * 16;
        *(uint4*)&b0 = *(const uint4*)w0;  *((uint4*)&b0 + 1) = *(const uint4*)(w0 + 8);
        *(uint4*)&b1 = *(const uint4*)w1;  *((uint4*)&b1 + 1) = *(const uint4*)(w1 + 8);
        *(uint4*)&b2 = *(const uint4*)w2;  *((uint4*)&b2 + 1) = *(const uint4*)(w2 + 8);
        *(uint4*)&b3 = *(const uint4*)w3;  *((uint4*)&b3 + 1) = *(const uint4*)(w3 + 8);
      }
      acc0 = __builtin_amdgcn_wmma_f32_16x16x32_f16(false, a, false, b0,
                                                    (short)0, acc0, false, false);
      acc1 = __builtin_amdgcn_wmma_f32_16x16x32_f16(false, a, false, b1,
                                                    (short)0, acc1, false, false);
      acc2 = __builtin_amdgcn_wmma_f32_16x16x32_f16(false, a, false, b2,
                                                    (short)0, acc2, false, false);
      acc3 = __builtin_amdgcn_wmma_f32_16x16x32_f16(false, a, false, b3,
                                                    (short)0, acc3, false, false);
    }
  }

#pragma unroll
  for (int j = 0; j < 4; ++j) {
    const v8f acc = (j == 0) ? acc0 : (j == 1) ? acc1 : (j == 2) ? acc2 : acc3;
    const int col = n0 + j * 16 + r;
    const float bv = bias[col];
#pragma unroll
    for (int i = 0; i < 8; ++i) {
      const int row = m0 + i + half * 8;
      out[(size_t)row * HID + col] = acc[i] + bv;
    }
  }
}

// ---------------- batch-norm ----------------

// per-column sum / sum-of-squares (HID == blockDim.x == 256)
__global__ void col_stats(const float* __restrict__ pre,
                          float* __restrict__ csum, float* __restrict__ csq,
                          int n) {
  const int c = threadIdx.x;
  float s = 0.0f, q = 0.0f;
  for (int row = blockIdx.x; row < n; row += gridDim.x) {
    const float v = pre[(size_t)row * HID + c];
    s += v;
    q += v * v;
  }
  atomicAdd(&csum[c], s);
  atomicAdd(&csq[c], q);
}

// normalize + ReLU, write f16 (feeds next layer's scatter + GEMM)
__global__ void bn_apply_relu_f16(const float* __restrict__ pre,
                                  const float* __restrict__ csum,
                                  const float* __restrict__ csq,
                                  const float* __restrict__ gamma,
                                  const float* __restrict__ beta,
                                  _Float16* __restrict__ dst, int n) {
  const int c = threadIdx.x;  // 0..255
  const float inv_n = 1.0f / (float)n;
  const float mu  = csum[c] * inv_n;
  const float var = csq[c] * inv_n - mu * mu;
  const float sc  = gamma[c] * rsqrtf(var + BN_EPS);
  const float sh  = beta[c] - mu * sc;
  for (int row = blockIdx.x; row < n; row += gridDim.x) {
    float v = pre[(size_t)row * HID + c] * sc + sh;
    dst[(size_t)row * HID + c] = (_Float16)fmaxf(v, 0.0f);
  }
}

// normalize, write f32 (final output, in place over pre)
__global__ void bn_apply_f32(const float* __restrict__ pre,
                             const float* __restrict__ csum,
                             const float* __restrict__ csq,
                             const float* __restrict__ gamma,
                             const float* __restrict__ beta,
                             float* __restrict__ dst, int n) {
  const int c = threadIdx.x;
  const float inv_n = 1.0f / (float)n;
  const float mu  = csum[c] * inv_n;
  const float var = csq[c] * inv_n - mu * mu;
  const float sc  = gamma[c] * rsqrtf(var + BN_EPS);
  const float sh  = beta[c] - mu * sc;
  for (int row = blockIdx.x; row < n; row += gridDim.x)
    dst[(size_t)row * HID + c] = pre[(size_t)row * HID + c] * sc + sh;
}

// ---------------- host-side orchestration ----------------

static inline size_t align256(size_t x) { return (x + 255) & ~(size_t)255; }

extern "C" void kernel_launch(void* const* d_in, const int* in_sizes, int n_in,
                              void* d_out, int out_size, void* d_ws,
                              size_t ws_size, hipStream_t stream) {
  (void)in_sizes; (void)n_in; (void)out_size; (void)ws_size;

  // inputs (setup_inputs order)
  const float* x    = (const float*)d_in[0];
  const int*   eidx = (const int*)d_in[1];          // [2, E]: src row, dst row
  const float* Wl0  = (const float*)d_in[2];
  const float* bl0  = (const float*)d_in[3];
  const float* Wr0  = (const float*)d_in[4];
  const float* g0   = (const float*)d_in[5];
  const float* be0  = (const float*)d_in[6];
  const float* Wl1  = (const float*)d_in[7];
  const float* bl1  = (const float*)d_in[8];
  const float* Wr1  = (const float*)d_in[9];
  const float* g1   = (const float*)d_in[10];
  const float* be1  = (const float*)d_in[11];
  const int* srcv = eidx;
  const int* dstv = eidx + N_EDGES;

  float* out = (float*)d_out;  // pre-BN scratch for both layers, final result

  // workspace layout (~116 MB)
  char* ws = (char*)d_ws;
  size_t off = 0;
  float* cnt = (float*)(ws + off);          off += align256((size_t)N_NODES * 4);
  float* sumbuf = (float*)(ws + off);       off += align256((size_t)N_NODES * HID * 4);
  _Float16* x16 = (_Float16*)(ws + off);    off += align256((size_t)N_NODES * IN_DIM * 2);
  _Float16* h16 = (_Float16*)(ws + off);    off += align256((size_t)N_NODES * HID * 2);
  _Float16* mean16 = (_Float16*)(ws + off); off += align256((size_t)N_NODES * HID * 2);
  float* csum = (float*)(ws + off);         off += align256((size_t)HID * 4);
  float* csq = (float*)(ws + off);          off += align256((size_t)HID * 4);
  _Float16* wl0h = (_Float16*)(ws + off);   off += align256((size_t)HID * IN_DIM * 2);
  _Float16* wr0h = (_Float16*)(ws + off);   off += align256((size_t)HID * IN_DIM * 2);
  _Float16* wl1h = (_Float16*)(ws + off);   off += align256((size_t)HID * HID * 2);
  _Float16* wr1h = (_Float16*)(ws + off);   off += align256((size_t)HID * HID * 2);

  const int T = 256;
  const int wavesPerBlock = T / 32;
  const int totalWaves = (N_NODES / 16) * (HID / 64);               // 12500
  const int gemmBlocks = (totalWaves + wavesPerBlock - 1) / wavesPerBlock;

  // weights + input features -> f16
  cvt_f32_to_f16<<<128, T, 0, stream>>>(Wl0, wl0h, HID * IN_DIM);
  cvt_f32_to_f16<<<128, T, 0, stream>>>(Wr0, wr0h, HID * IN_DIM);
  cvt_f32_to_f16<<<256, T, 0, stream>>>(Wl1, wl1h, HID * HID);
  cvt_f32_to_f16<<<256, T, 0, stream>>>(Wr1, wr1h, HID * HID);
  cvt_f32_to_f16<<<2048, T, 0, stream>>>(x, x16, N_NODES * IN_DIM);

  // ---- layer 0 ----
  zero_f32<<<64, T, 0, stream>>>(cnt, N_NODES);
  zero_f32<<<2048, T, 0, stream>>>(sumbuf, (long long)N_NODES * IN_DIM);
  zero_f32<<<1, T, 0, stream>>>(csum, HID);
  zero_f32<<<1, T, 0, stream>>>(csq, HID);
  count_deg<<<1024, T, 0, stream>>>(dstv, cnt, N_EDGES);
  scatter_sum_f16<<<4096, T, 0, stream>>>(x16, srcv, dstv, sumbuf, N_EDGES, IN_DIM);
  mean_cvt<<<2048, T, 0, stream>>>(sumbuf, cnt, mean16, N_NODES, IN_DIM);
  sage_gemm_wmma<<<gemmBlocks, T, 0, stream>>>(mean16, x16, wl0h, wr0h, bl0,
                                               out, N_NODES, IN_DIM);
  col_stats<<<1024, T, 0, stream>>>(out, csum, csq, N_NODES);
  bn_apply_relu_f16<<<1024, T, 0, stream>>>(out, csum, csq, g0, be0, h16, N_NODES);

  // ---- layer 1 ----
  zero_f32<<<4096, T, 0, stream>>>(sumbuf, (long long)N_NODES * HID);
  zero_f32<<<1, T, 0, stream>>>(csum, HID);
  zero_f32<<<1, T, 0, stream>>>(csq, HID);
  scatter_sum_f16<<<8192, T, 0, stream>>>(h16, srcv, dstv, sumbuf, N_EDGES, HID);
  mean_cvt<<<4096, T, 0, stream>>>(sumbuf, cnt, mean16, N_NODES, HID);
  sage_gemm_wmma<<<gemmBlocks, T, 0, stream>>>(mean16, h16, wl1h, wr1h, bl1,
                                               out, N_NODES, HID);
  col_stats<<<1024, T, 0, stream>>>(out, csum, csq, N_NODES);
  bn_apply_f32<<<1024, T, 0, stream>>>(out, csum, csq, g1, be1, out, N_NODES);
}
